// MultiHeadSelfAttention_21921513079123
// MI455X (gfx1250) — matrix operand order repaired and checked by reading, not compiled, not run on hardware
//
#include <hip/hip_runtime.h>

// ---------------------------------------------------------------------------
// Multi-head self-attention for MI455X (gfx1250), bf16 WMMA + fp32 accumulate.
// b=4, n=2048, f=1024, h=16, hd=64.
// Round 2: 32x64 wave tiles in GEMM (8 WMMA per K-step per wave) and
// GLOBAL_LOAD_ASYNC_TO_LDS_B128 for direct global->LDS tile staging.
// ---------------------------------------------------------------------------

#define NTOK  2048
#define FEATS 1024
#define NB    4
#define NH    16
#define HD    64

typedef __attribute__((ext_vector_type(16))) __bf16 v16bf;
typedef __attribute__((ext_vector_type(8)))  float  v8f;

union FragBF {
  unsigned int u[8];
  v16bf v;
};

__device__ __forceinline__ unsigned short f2bf(float f) {
  unsigned int u = __float_as_uint(f);
  u += 0x7FFFu + ((u >> 16) & 1u);       // round-to-nearest-even
  return (unsigned short)(u >> 16);
}
__device__ __forceinline__ unsigned int pack2bf(float a, float b) {
  return (unsigned int)f2bf(a) | ((unsigned int)f2bf(b) << 16);
}

__device__ __forceinline__ v8f wmma_bf16(const FragBF& a, const FragBF& b, v8f c) {
  // D = A(16x32 bf16) x B(32x16 bf16) + C(16x16 f32)
  return __builtin_amdgcn_wmma_f32_16x16x32_bf16(
      /*neg_a=*/false, a.v, /*neg_b=*/false, b.v,
      /*c_mod=*/(short)0, c, /*reuse_a=*/false, /*reuse_b=*/false);
}

// LDS offset of a __shared__ element: generic LDS pointers carry the
// workgroup-relative offset in the low 32 bits; hardware adds LDS_BASE.
__device__ __forceinline__ unsigned lds_off(const void* p) {
  return (unsigned)(uintptr_t)p;
}

// Async 16-byte global -> LDS copy (per lane), tracked by ASYNCcnt.
__device__ __forceinline__ void async_copy_b128(unsigned lds, const void* gptr) {
  unsigned long long ga = (unsigned long long)(uintptr_t)gptr;
  asm volatile("global_load_async_to_lds_b128 %0, %1, off"
               :: "v"(lds), "v"(ga) : "memory");
}
__device__ __forceinline__ void wait_async() {
  asm volatile("s_wait_asynccnt 0x0" ::: "memory");
}

// ---------------------------------------------------------------------------
// GEMM: C[M,N] = A[M,K] @ B[K,N] + bias   (M=8192, N=K=1024)
// ABF16: 0 = A is fp32 (converted to bf16 while staging),
//        1 = A is bf16 raw (staged with async global->LDS B128 copies).
// OMODE: 0 = write bf16 with (b,n,f)->(b,h,n,hd) permute (QKV projections),
//        1 = write fp32 row-major (final output).
// Block: 256 threads (8 waves). Tile: 256x64, K-step 32. Each wave: 32x64.
// ---------------------------------------------------------------------------
template <int ABF16, int OMODE>
__global__ __launch_bounds__(256) void gemm_kernel(const void* __restrict__ Aptr,
                                                   const float* __restrict__ B,
                                                   const float* __restrict__ bias,
                                                   void* __restrict__ Cptr) {
  constexpr int N = FEATS;
  constexpr int K = FEATS;
  constexpr int LDA = 40;   // ushort stride: 80 B rows, 16B-aligned, bank-clean
  constexpr int LDB = 34;   // padded transposed weight tile

  __shared__ unsigned short As[256 * LDA];   // A tile, bf16, row-major
  __shared__ unsigned short Bt[64 * LDB];    // B tile, bf16, transposed [n][k]

  const int tid  = threadIdx.x;
  const int wave = tid >> 5;
  const int lane = tid & 31;
  const int hl   = lane & 15;
  const int hi   = lane >> 4;
  const int m0   = blockIdx.y * 256;
  const int n0   = blockIdx.x * 64;

  v8f acc[2][4] = {};

  const int arow = tid;                 // 0..255, one row per thread
  const int brow = tid >> 3;            // 0..31 (k within tile)
  const int bnx  = (tid & 7) * 8;       // 0..56

  for (int k0 = 0; k0 < K; k0 += 32) {
    // ---- stage A tile: one full 32-element row per thread ----
    if (ABF16) {
      const unsigned short* Ab = (const unsigned short*)Aptr;
      const unsigned short* g  = Ab + (size_t)(m0 + arow) * K + k0;
      const unsigned ldst = lds_off(&As[arow * LDA]);
#pragma unroll
      for (int c = 0; c < 4; ++c)
        async_copy_b128(ldst + 16 * c, g + 8 * c);
    } else {
      const float* Af = (const float*)Aptr;
      const float4* s4 = (const float4*)(Af + (size_t)(m0 + arow) * K + k0);
      unsigned int* dst = (unsigned int*)&As[arow * LDA];
#pragma unroll
      for (int c = 0; c < 4; ++c) {
        const float4 fa = s4[2 * c], fb = s4[2 * c + 1];
        dst[4 * c + 0] = pack2bf(fa.x, fa.y);
        dst[4 * c + 1] = pack2bf(fa.z, fa.w);
        dst[4 * c + 2] = pack2bf(fb.x, fb.y);
        dst[4 * c + 3] = pack2bf(fb.z, fb.w);
      }
    }
    // ---- stage B tile transposed (weights are fp32) ----
    {
      const float4* s4 = (const float4*)(B + (size_t)(k0 + brow) * N + n0 + bnx);
      const float4 f0 = s4[0], f1 = s4[1];
      const float vv[8] = {f0.x, f0.y, f0.z, f0.w, f1.x, f1.y, f1.z, f1.w};
#pragma unroll
      for (int e = 0; e < 8; ++e)
        Bt[(bnx + e) * LDB + brow] = f2bf(vv[e]);
    }
    if (k0 + 32 < K)   // prefetch next weight tile -> global_prefetch_b8
      __builtin_prefetch(B + (size_t)(k0 + 32 + brow) * N + n0 + bnx, 0, 1);
    if (ABF16) wait_async();
    __syncthreads();

    // ---- compute: 2 A-frags x 4 B-frags = 8 WMMAs per wave ----
    FragBF af[2];
#pragma unroll
    for (int a = 0; a < 2; ++a) {
      const int row = 32 * wave + 16 * a + hl;
#pragma unroll
      for (int g = 0; g < 8; ++g) {
        const int kb = 2 * g + ((g >= 4) ? 8 : 0) + 8 * hi;  // A layout K pair
        af[a].u[g] = *(const unsigned int*)&As[row * LDA + kb];
      }
    }
#pragma unroll
    for (int t = 0; t < 4; ++t) {
      FragBF bf;
      const int nn = 16 * t + hl;
#pragma unroll
      for (int g = 0; g < 8; ++g) {
        const int kk = 2 * g + ((g >= 4) ? 8 : 0) + 8 * hi;  // B layout K pair
        bf.u[g] = *(const unsigned int*)&Bt[nn * LDB + kk];
      }
      acc[0][t] = wmma_bf16(af[0], bf, acc[0][t]);
      acc[1][t] = wmma_bf16(af[1], bf, acc[1][t]);
    }
    __syncthreads();
  }

  // ---- epilogue: C layout row M = r + 8*hi (VGPR r), col N = 16t + hl ----
#pragma unroll
  for (int a = 0; a < 2; ++a)
#pragma unroll
    for (int t = 0; t < 4; ++t) {
      const int col = n0 + 16 * t + hl;
      const float bv = bias[col];
#pragma unroll
      for (int r = 0; r < 8; ++r) {
        const int row = m0 + 32 * wave + 16 * a + r + 8 * hi;
        const float val = acc[a][t][r] + bv;
        if (OMODE == 0) {
          unsigned short* Cb = (unsigned short*)Cptr;
          const int b_idx = row >> 11;          // / NTOK
          const int i     = row & (NTOK - 1);
          const int head  = col >> 6;           // / HD
          const int d     = col & (HD - 1);
          Cb[(((size_t)(b_idx * NH + head)) * NTOK + i) * HD + d] = f2bf(val);
        } else {
          float* Cf = (float*)Cptr;
          Cf[(size_t)row * N + col] = val;
        }
      }
    }
}

// ---------------------------------------------------------------------------
// Flash-attention: grid (n/128, b*h), 256 threads (8 waves).
// Wave w owns 16 query rows; workgroup shares K/V staging for 32-key blocks.
// K tile staged with async global->LDS B128; V staged transposed (VALU).
// logits = (Q.K^T)/32 + m_i*m_j  (additive {0,1} mask), online softmax,
// O accumulated in fp32, final /l, stored bf16 as (b, i, h*64+d).
// ---------------------------------------------------------------------------
__global__ __launch_bounds__(256) void attn_kernel(const unsigned short* __restrict__ Qb,
                                                   const unsigned short* __restrict__ Kb,
                                                   const unsigned short* __restrict__ Vb,
                                                   const int* __restrict__ mask,
                                                   unsigned short* __restrict__ attnb) {
  constexpr int LDK = 72;   // K tile [32][64]: 144 B rows, 16B-aligned, bank-clean
  constexpr int LDV = 34;   // V tile transposed [64][32] padded
  constexpr int LDP = 34;   // P scratch per-wave [16][32] padded

  __shared__ unsigned short Ks[32 * LDK];
  __shared__ unsigned short Vt[64 * LDV];
  __shared__ unsigned short Ps[8 * 16 * LDP];

  const int tid  = threadIdx.x;
  const int wave = tid >> 5;
  const int lane = tid & 31;
  const int hl   = lane & 15;
  const int hi   = lane >> 4;
  const int bh    = blockIdx.y;
  const int b_idx = bh >> 4;
  const int h_idx = bh & 15;
  const int i0    = blockIdx.x * 128 + wave * 16;

  const size_t qkv_base = (size_t)bh * NTOK * HD;

  // Q fragments: 16 rows x 64 d  -> two 16x32 A-frags, loaded once.
  FragBF qa[2];
#pragma unroll
  for (int f = 0; f < 2; ++f)
#pragma unroll
    for (int g = 0; g < 8; ++g) {
      const int kb = 2 * g + ((g >= 4) ? 8 : 0) + 8 * hi + 32 * f;
      qa[f].u[g] = *(const unsigned int*)&Qb[qkv_base + (size_t)(i0 + hl) * HD + kb];
    }

  float mi[8];
#pragma unroll
  for (int r = 0; r < 8; ++r)
    mi[r] = (float)mask[b_idx * NTOK + i0 + r + 8 * hi];

  float m_r[8], l_r[8];
  v8f o[4] = {};
#pragma unroll
  for (int r = 0; r < 8; ++r) { m_r[r] = -3.0e38f; l_r[r] = 0.0f; }

  const int krow = tid >> 3;        // 0..31 (key row in tile)
  const int kdx  = (tid & 7) * 8;   // 0..56 (d offset, 16B granules)

  for (int j0 = 0; j0 < NTOK; j0 += 32) {
    // ---- stage K via async B128, V transposed via VALU ----
    {
      async_copy_b128(lds_off(&Ks[krow * LDK + kdx]),
                      &Kb[qkv_base + (size_t)(j0 + krow) * HD + kdx]);

      union { uint4 q; unsigned short s[8]; } uv;
      uv.q = *(const uint4*)&Vb[qkv_base + (size_t)(j0 + krow) * HD + kdx];
#pragma unroll
      for (int e = 0; e < 8; ++e)
        Vt[(kdx + e) * LDV + krow] = uv.s[e];
    }
    if (j0 + 32 < NTOK)
      __builtin_prefetch(&Vb[qkv_base + (size_t)(j0 + 32 + krow) * HD + kdx], 0, 1);
    wait_async();
    __syncthreads();

    // ---- logits 16x32: two 16x16 tiles, each = 2 WMMAs over d ----
    v8f c0 = {}, c1 = {};
#pragma unroll
    for (int f = 0; f < 2; ++f) {
      FragBF kf;
#pragma unroll
      for (int g = 0; g < 8; ++g)
        kf.u[g] = *(const unsigned int*)&Ks[hl * LDK + 2 * g + ((g >= 4) ? 8 : 0) + 8 * hi + 32 * f];
      c0 = wmma_bf16(qa[f], kf, c0);
    }
#pragma unroll
    for (int f = 0; f < 2; ++f) {
      FragBF kf;
#pragma unroll
      for (int g = 0; g < 8; ++g)
        kf.u[g] = *(const unsigned int*)&Ks[(16 + hl) * LDK + 2 * g + ((g >= 4) ? 8 : 0) + 8 * hi + 32 * f];
      c1 = wmma_bf16(qa[f], kf, c1);
    }

    const float mj0 = (float)mask[b_idx * NTOK + j0 + hl];
    const float mj1 = (float)mask[b_idx * NTOK + j0 + 16 + hl];

    // ---- online softmax (rows live per-VGPR; cols across 16-lane halves) ----
    float alpha[8];
    constexpr float scale = 0.03125f;  // 1/sqrt(1024)
#pragma unroll
    for (int r = 0; r < 8; ++r) {
      const float v0 = c0[r] * scale + mi[r] * mj0;
      const float v1 = c1[r] * scale + mi[r] * mj1;
      float rm = fmaxf(v0, v1);
#pragma unroll
      for (int off = 1; off < 16; off <<= 1)
        rm = fmaxf(rm, __shfl_xor(rm, off, 32));
      const float mn = fmaxf(m_r[r], rm);
      alpha[r] = __expf(m_r[r] - mn);
      m_r[r] = mn;
      const float p0 = __expf(v0 - mn);
      const float p1 = __expf(v1 - mn);
      float rs = p0 + p1;
#pragma unroll
      for (int off = 1; off < 16; off <<= 1)
        rs += __shfl_xor(rs, off, 32);
      l_r[r] = l_r[r] * alpha[r] + rs;

      const int prow = r + 8 * hi;
      Ps[(wave * 16 + prow) * LDP + hl]      = f2bf(p0);
      Ps[(wave * 16 + prow) * LDP + 16 + hl] = f2bf(p1);
    }
#pragma unroll
    for (int t = 0; t < 4; ++t)
#pragma unroll
      for (int r = 0; r < 8; ++r)
        o[t][r] *= alpha[r];

    asm volatile("" ::: "memory");  // keep P stores before A-frag reloads

    // ---- P (C layout) -> A-frag via LDS, then P.V: 4 WMMAs ----
    FragBF pf;
#pragma unroll
    for (int g = 0; g < 8; ++g) {
      const int kb = 2 * g + ((g >= 4) ? 8 : 0) + 8 * hi;
      pf.u[g] = *(const unsigned int*)&Ps[(wave * 16 + hl) * LDP + kb];
    }
#pragma unroll
    for (int t = 0; t < 4; ++t) {
      FragBF vf;
#pragma unroll
      for (int g = 0; g < 8; ++g)
        vf.u[g] = *(const unsigned int*)&Vt[(16 * t + hl) * LDV + 2 * g + ((g >= 4) ? 8 : 0) + 8 * hi];
      o[t] = wmma_bf16(pf, vf, o[t]);
    }
    __syncthreads();
  }

  // ---- normalize and store attn (b, i, h*64+d) as bf16 ----
#pragma unroll
  for (int t = 0; t < 4; ++t)
#pragma unroll
    for (int r = 0; r < 8; ++r) {
      const int row = i0 + r + 8 * hi;
      const int d   = 16 * t + hl;
      const float val = o[t][r] / l_r[r];
      attnb[((size_t)(b_idx * NTOK) + row) * FEATS + h_idx * HD + d] = f2bf(val);
    }
}

// ---------------------------------------------------------------------------
extern "C" void kernel_launch(void* const* d_in, const int* in_sizes, int n_in,
                              void* d_out, int out_size, void* d_ws, size_t ws_size,
                              hipStream_t stream) {
  (void)in_sizes; (void)n_in; (void)out_size; (void)ws_size;
  const float* x    = (const float*)d_in[0];
  const int*   mask = (const int*)d_in[1];
  const float* Wq   = (const float*)d_in[2];
  const float* bq   = (const float*)d_in[3];
  const float* Wk   = (const float*)d_in[4];
  const float* bk   = (const float*)d_in[5];
  const float* Wv   = (const float*)d_in[6];
  const float* bv   = (const float*)d_in[7];
  const float* Wo   = (const float*)d_in[8];
  const float* bo   = (const float*)d_in[9];
  float* out = (float*)d_out;

  // Workspace (bf16 raw ushort): Q, K, V in (b*h, n, hd), attn in (b*n, f).
  const size_t qkv_elems = (size_t)NB * NH * NTOK * HD;   // 8,388,608
  unsigned short* Qb = (unsigned short*)d_ws;
  unsigned short* Kb = Qb + qkv_elems;
  unsigned short* Vb = Kb + qkv_elems;
  unsigned short* Ab = Vb + qkv_elems;                    // (b*n, f) bf16

  dim3 blk(256);
  dim3 ggrid(FEATS / 64, (NB * NTOK) / 256);   // (16, 32)

  gemm_kernel<0, 0><<<ggrid, blk, 0, stream>>>(x, Wq, bq, Qb);
  gemm_kernel<0, 0><<<ggrid, blk, 0, stream>>>(x, Wk, bk, Kb);
  gemm_kernel<0, 0><<<ggrid, blk, 0, stream>>>(x, Wv, bv, Vb);

  dim3 agrid(NTOK / 128, NB * NH);             // (16, 64)
  attn_kernel<<<agrid, blk, 0, stream>>>(Qb, Kb, Vb, mask, Ab);

  gemm_kernel<1, 1><<<ggrid, blk, 0, stream>>>(Ab, Wo, bo, out);
}
